// ExecuTorchModel_65944927863403
// MI455X (gfx1250) — compile-verified, matrix operand order checked by reading, not run on hardware
//
#include <hip/hip_runtime.h>
#include <hip/hip_bf16.h>

// Problem constants (match reference)
#define NN 32768
#define KK 16
#define EE (NN * KK)
#define HH 128
#define H2 256
#define RRR 2
#define LLL 2

typedef __attribute__((ext_vector_type(16))) __bf16 v16bf;
typedef __attribute__((ext_vector_type(8)))  __bf16 v8bf;
typedef __attribute__((ext_vector_type(8)))  float  v8f;

// ---------------------------------------------------------------------------
// WMMA helpers (CDNA5: D = A(16x32 bf16) x B(32x16 bf16) + C(16x16 f32))
// ---------------------------------------------------------------------------
__device__ __forceinline__ v8f wmma_bf16(v16bf a, v16bf b, v8f c) {
    return __builtin_amdgcn_wmma_f32_16x16x32_bf16(
        /*neg_a=*/false, a, /*neg_b=*/false, b,
        /*c_mod=*/(short)0, c, /*reuse_a=*/false, /*reuse_b=*/false);
}

// A fragment (16x32 bf16) from fp32 row-major [*, ld].
// Per ISA layout: lanes 0-15 -> M=lane, K in {kg..kg+7, 16+kg..16+kg+7},
// kg = (lane>>4)*8; VGPR j holds K pair (2j,2j+1) within group.
__device__ __forceinline__ v16bf load_a_frag_f32(const float* A, int ld,
                                                 int row0, int k0) {
    int lane = threadIdx.x & 31;
    int m    = lane & 15;
    int kg   = (lane >> 4) << 3;
    const float* p = A + (size_t)(row0 + m) * ld + k0 + kg;
    v16bf a;
#pragma unroll
    for (int j = 0; j < 4; ++j) {
        float2 lo = *(const float2*)(p + 2 * j);
        float2 hi = *(const float2*)(p + 16 + 2 * j);
        a[2 * j + 0]     = (__bf16)lo.x;
        a[2 * j + 1]     = (__bf16)lo.y;
        a[8 + 2 * j + 0] = (__bf16)hi.x;
        a[8 + 2 * j + 1] = (__bf16)hi.y;
    }
    return a;
}

// A fragment from bf16 row-major [*, ld] (two 16B contiguous chunks per lane).
__device__ __forceinline__ v16bf load_a_frag_bf16(const __bf16* A, int ld,
                                                  int row0, int k0) {
    int lane = threadIdx.x & 31;
    int m    = lane & 15;
    int kg   = (lane >> 4) << 3;
    const __bf16* p = A + (size_t)(row0 + m) * ld + k0 + kg;
    v8bf lo = *(const v8bf*)(p);
    v8bf hi = *(const v8bf*)(p + 16);
    v16bf a;
#pragma unroll
    for (int j = 0; j < 8; ++j) { a[j] = lo[j]; a[8 + j] = hi[j]; }
    return a;
}

// B fragment from pre-swizzled weights: tile = 512 bf16, lane owns 16
// contiguous bf16 (one 32-byte load).
__device__ __forceinline__ v16bf load_b_frag(const __bf16* Bsw, int tile) {
    int lane = threadIdx.x & 31;
    return *(const v16bf*)(Bsw + ((size_t)tile << 9) + (lane << 4));
}

// ---------------------------------------------------------------------------
// Prep: pack W (row-major [Ncol, Kdim], B[k][n] = W[n][k]) into swizzled bf16
// fragment layout. Tile t = kt*(Ncol/16)+nt; element i = t*512 + lane*16 + j.
// lane -> n = nt*16 + (lane&15), kg = (lane>>4)*8;
// j<8 -> k = kt*32 + kg + j;  j>=8 -> k = kt*32 + 16 + kg + (j-8).
// ---------------------------------------------------------------------------
__global__ void pack_b_kernel(const float* __restrict__ W, __bf16* __restrict__ Bsw,
                              int Kdim, int Ncol) {
    int i = blockIdx.x * blockDim.x + threadIdx.x;
    int ntilesN = Ncol >> 4;
    int total = (Kdim >> 5) * ntilesN * 512;
    if (i >= total) return;
    int j    = i & 15;
    int lane = (i >> 4) & 31;
    int t    = i >> 9;
    int nt   = t % ntilesN;
    int kt   = t / ntilesN;
    int n    = nt * 16 + (lane & 15);
    int kg   = (lane >> 4) << 3;
    int k    = kt * 32 + ((j < 8) ? (kg + j) : (16 + kg + (j - 8)));
    Bsw[i] = (__bf16)W[(size_t)n * Kdim + k];
}

__global__ void bn_prep_kernel(const float* __restrict__ g, const float* __restrict__ b,
                               const float* __restrict__ m, const float* __restrict__ v,
                               float* __restrict__ scale, float* __restrict__ shift, int n) {
    int i = blockIdx.x * blockDim.x + threadIdx.x;
    if (i >= n) return;
    float sc = g[i] * rsqrtf(v[i] + 1e-5f);
    scale[i] = sc;
    shift[i] = b[i] - m[i] * sc;
}

// ---------------------------------------------------------------------------
// Fused message + per-node softmax + aggregate + residual.
// One wave32 per node, 4 channels per lane (float4). Edges of node n are
// nbr[n,k] (edge ids, -1 = pad); src gather rows fully coalesced (512B/row).
// ---------------------------------------------------------------------------
__global__ void __launch_bounds__(256) agg_kernel(
    const float* __restrict__ x, const int* __restrict__ src,
    const float* __restrict__ ea, const int* __restrict__ nbr,
    const float* __restrict__ We, float* __restrict__ out) {
    int wave = threadIdx.x >> 5;
    int lane = threadIdx.x & 31;
    int node = blockIdx.x * 8 + wave;
    if (node >= NN) return;
    int c = lane << 2;

    const float4 xn = *(const float4*)(x + (size_t)node * HH + c);
    const float4 we = *(const float4*)(We + c);

    float mr[4], sr[4], ar[4];
#pragma unroll
    for (int t = 0; t < 4; ++t) { mr[t] = -3.0e38f; sr[t] = 0.f; ar[t] = 0.f; }

    const int* nb = nbr + (size_t)node * KK;
#pragma unroll
    for (int k = 0; k < KK; ++k) {
        int e = nb[k];
        if (e < 0) continue;
        float eav = ea[e];
        int   si  = src[e];
        const float4 xj = *(const float4*)(x + (size_t)si * HH + c);
        float gv[4];
        gv[0] = fmaxf(fmaf(eav, we.x, xj.x), 0.f) + 1e-7f;
        gv[1] = fmaxf(fmaf(eav, we.y, xj.y), 0.f) + 1e-7f;
        gv[2] = fmaxf(fmaf(eav, we.z, xj.z), 0.f) + 1e-7f;
        gv[3] = fmaxf(fmaf(eav, we.w, xj.w), 0.f) + 1e-7f;
#pragma unroll
        for (int t = 0; t < 4; ++t) {  // online softmax accumulation
            float nm = fmaxf(mr[t], gv[t]);
            float so = __expf(mr[t] - nm);
            float p  = __expf(gv[t] - nm);
            sr[t] = sr[t] * so + p;
            ar[t] = ar[t] * so + gv[t] * p;
            mr[t] = nm;
        }
    }
    float4 o;
    o.x = ar[0] / (sr[0] + 1e-16f) + xn.x;
    o.y = ar[1] / (sr[1] + 1e-16f) + xn.y;
    o.z = ar[2] / (sr[2] + 1e-16f) + xn.z;
    o.w = ar[3] / (sr[3] + 1e-16f) + xn.w;
    *(float4*)(out + (size_t)node * HH + c) = o;
}

// ---------------------------------------------------------------------------
// GEMM1: H[N,256] = relu(bn(A[N,128] @ B1))  -> bf16 output.
// 8 waves/block, 16 rows/wave, A fragments register-resident, K loop = 4.
// ---------------------------------------------------------------------------
__global__ void __launch_bounds__(256) gemm1_kernel(
    const float* __restrict__ A, const __bf16* __restrict__ Bsw,
    const float* __restrict__ scale, const float* __restrict__ shift,
    __bf16* __restrict__ Hout) {
    int wave = threadIdx.x >> 5;
    int lane = threadIdx.x & 31;
    int row0 = blockIdx.x * 128 + wave * 16;

    v16bf a[4];
#pragma unroll
    for (int kk = 0; kk < 4; ++kk) a[kk] = load_a_frag_f32(A, HH, row0, kk * 32);

    int col_lo = lane & 15;
    int rbase  = row0 + ((lane >> 4) << 3);  // C/D: lanes 16-31 hold M=8..15
#pragma unroll
    for (int nt = 0; nt < 16; ++nt) {
        v8f c = {};
#pragma unroll
        for (int kk = 0; kk < 4; ++kk)
            c = wmma_bf16(a[kk], load_b_frag(Bsw, kk * 16 + nt), c);
        int col = nt * 16 + col_lo;
        float sc = scale[col], sh = shift[col];
#pragma unroll
        for (int rr = 0; rr < 8; ++rr) {
            float h = fmaf(c[rr], sc, sh);
            h = h > 0.f ? h : 0.f;
            Hout[(size_t)(rbase + rr) * H2 + col] = (__bf16)h;
        }
    }
}

// ---------------------------------------------------------------------------
// GEMM2: Y[N,128] (+)= H[N,256] @ B2, optional LeakyReLU on final accumulate.
// ---------------------------------------------------------------------------
__global__ void __launch_bounds__(256) gemm2_kernel(
    const __bf16* __restrict__ A, const __bf16* __restrict__ Bsw,
    float* __restrict__ Y, int accumulate, int leaky) {
    int wave = threadIdx.x >> 5;
    int lane = threadIdx.x & 31;
    int row0 = blockIdx.x * 128 + wave * 16;

    v16bf a[8];
#pragma unroll
    for (int kk = 0; kk < 8; ++kk) a[kk] = load_a_frag_bf16(A, H2, row0, kk * 32);

    int col_lo = lane & 15;
    int rbase  = row0 + ((lane >> 4) << 3);
#pragma unroll
    for (int nt = 0; nt < 8; ++nt) {
        v8f c = {};
#pragma unroll
        for (int kk = 0; kk < 8; ++kk)
            c = wmma_bf16(a[kk], load_b_frag(Bsw, kk * 8 + nt), c);
        int col = nt * 16 + col_lo;
#pragma unroll
        for (int rr = 0; rr < 8; ++rr) {
            size_t o = (size_t)(rbase + rr) * HH + col;
            float y = c[rr];
            if (accumulate) y += Y[o];
            if (leaky) y = y > 0.f ? y : 0.01f * y;
            Y[o] = y;
        }
    }
}

// ---------------------------------------------------------------------------
// Host orchestration (graph-capture safe: only async launches on `stream`).
// ---------------------------------------------------------------------------
extern "C" void kernel_launch(void* const* d_in, const int* in_sizes, int n_in,
                              void* d_out, int out_size, void* d_ws, size_t ws_size,
                              hipStream_t stream) {
    (void)in_sizes; (void)n_in; (void)out_size; (void)ws_size;

    const float* x          = (const float*)d_in[0];
    const int*   edge_inds  = (const int*)d_in[1];    // [R,2,E]
    const float* edge_attrs = (const float*)d_in[2];  // [R,E,1]
    const int*   nbrs       = (const int*)d_in[3];    // [R,N,K]
    const float* W_edge     = (const float*)d_in[4];  // [L,R,H,1]
    const float* W1         = (const float*)d_in[5];  // [L,R,2H,H]
    const float* bng        = (const float*)d_in[6];
    const float* bnb        = (const float*)d_in[7];
    const float* bnm        = (const float*)d_in[8];
    const float* bnv        = (const float*)d_in[9];
    const float* W2         = (const float*)d_in[10]; // [L,R,H,2H]

    // Workspace carve-up (~48.5 MB)
    char* ws = (char*)d_ws;
    float*  out_agg = (float*)ws;   ws += (size_t)NN * HH * sizeof(float);     // 16 MB
    __bf16* hbuf    = (__bf16*)ws;  ws += (size_t)NN * H2 * sizeof(__bf16);    // 16 MB
    float*  ybuf    = (float*)ws;   ws += (size_t)NN * HH * sizeof(float);     // 16 MB
    __bf16* B1sw    = (__bf16*)ws;  ws += (size_t)LLL * RRR * HH * H2 * sizeof(__bf16);
    __bf16* B2sw    = (__bf16*)ws;  ws += (size_t)LLL * RRR * HH * H2 * sizeof(__bf16);
    float*  bnscale = (float*)ws;   ws += (size_t)LLL * RRR * H2 * sizeof(float);
    float*  bnshift = (float*)ws;   ws += (size_t)LLL * RRR * H2 * sizeof(float);

    // Weight / BN prep (tiny)
    const int packN = HH * H2;  // 32768 swizzled bf16 per matrix
    for (int lr = 0; lr < LLL * RRR; ++lr) {
        pack_b_kernel<<<(packN + 255) / 256, 256, 0, stream>>>(
            W1 + (size_t)lr * H2 * HH, B1sw + (size_t)lr * packN, HH, H2);
        pack_b_kernel<<<(packN + 255) / 256, 256, 0, stream>>>(
            W2 + (size_t)lr * HH * H2, B2sw + (size_t)lr * packN, H2, HH);
    }
    bn_prep_kernel<<<(LLL * RRR * H2 + 255) / 256, 256, 0, stream>>>(
        bng, bnb, bnm, bnv, bnscale, bnshift, LLL * RRR * H2);

    const float* xin = x;
    for (int l = 0; l < LLL; ++l) {
        float* yout = (l == LLL - 1) ? (float*)d_out : ybuf;
        for (int r = 0; r < RRR; ++r) {
            int lr = l * RRR + r;
            const int*   srcp = edge_inds + (size_t)r * 2 * EE;   // ei[r][0]
            const float* eap  = edge_attrs + (size_t)r * EE;
            const int*   nbp  = nbrs + (size_t)r * NN * KK;
            agg_kernel<<<NN / 8, 256, 0, stream>>>(
                xin, srcp, eap, nbp, W_edge + (size_t)lr * HH, out_agg);
            gemm1_kernel<<<NN / 128, 256, 0, stream>>>(
                out_agg, B1sw + (size_t)lr * packN,
                bnscale + (size_t)lr * H2, bnshift + (size_t)lr * H2, hbuf);
            gemm2_kernel<<<NN / 128, 256, 0, stream>>>(
                hbuf, B2sw + (size_t)lr * packN, yout,
                /*accumulate=*/(r > 0), /*leaky=*/(r == RRR - 1 && l < LLL - 1));
        }
        xin = yout;
    }
}
